// Router_49778670960792
// MI455X (gfx1250) — compile-verified, matrix operand order checked by reading, not run on hardware
//
#include <hip/hip_runtime.h>
#include <math.h>

typedef __bf16 bf16;
typedef __attribute__((ext_vector_type(16))) __bf16 v16bf;
typedef __attribute__((ext_vector_type(8)))  __bf16 v8bf;
typedef __attribute__((ext_vector_type(4)))  __bf16 v4bf;
typedef __attribute__((ext_vector_type(8)))  float  v8f;

#define D_    4096
#define H_    16
#define HD_   256
#define LV_   2048
#define LT_   1024
#define MLPH_ 8192

enum { EPI_BIAS = 1, EPI_RES = 2, EPI_GELU = 4, EPI_MASK = 8 };

typedef __attribute__((address_space(3))) bf16 lds_bf16;

// ---- CDNA5 async copy: global -> LDS (ASYNCcnt-tracked, no VGPR staging) ----
__device__ __forceinline__ void async_copy_b128(const bf16* g, bf16* l) {
  lds_bf16* lp = (lds_bf16*)l;   // 32-bit LDS address operand (VDST)
  asm volatile("global_load_async_to_lds_b128 %0, %1, off"
               :: "v"(lp), "v"(g) : "memory");
}

__device__ __forceinline__ void wait_async0() {
#if __has_builtin(__builtin_amdgcn_s_wait_asynccnt)
  __builtin_amdgcn_s_wait_asynccnt(0);
#else
  asm volatile("s_wait_asynccnt 0x0" ::: "memory");
#endif
}

// ---------------- fp32 -> bf16 convert ----------------
__global__ __launch_bounds__(256) void cvt_f32_bf16(const float* __restrict__ s,
                                                    bf16* __restrict__ d, long long n) {
  long long i = ((long long)blockIdx.x * 256 + threadIdx.x) * 4;
  if (i >= n) return;
  const float4 f = *(const float4*)(s + i);
  v4bf o;
  o[0] = (bf16)f.x; o[1] = (bf16)f.y; o[2] = (bf16)f.z; o[3] = (bf16)f.w;
  *(v4bf*)(d + i) = o;
}

// ---------------- generic bf16 WMMA GEMM:  C = A(MxK) * W(NxK)^T  ----------------
// 256 threads = 8 waves (2x4), block tile 128x128, K-step 32, double-buffered LDS
// filled by global_load_async_to_lds_b128. LDS pitch 40 halves (80B = 20 banks)
// -> conflict-free b128 fragment reads.
template<int EPI, typename OutT>
__global__ __launch_bounds__(256) void gemm_bf16_kernel(
    const bf16* __restrict__ A, const bf16* __restrict__ W, OutT* __restrict__ C,
    int M, int N, int K, int lda, int ldw, int ldc,
    long long aBatch, long long wBatch, long long cBatch,
    const float* __restrict__ bias, const float* __restrict__ res, int ldres,
    const float* __restrict__ mask, float scale)
{
  __shared__ bf16 As[2][128 * 40];
  __shared__ bf16 Ws[2][128 * 40];
  const int tid = threadIdx.x;
  const int bm = blockIdx.y, bn = blockIdx.x, bz = blockIdx.z;
  A += (long long)bz * aBatch;
  W += (long long)bz * wBatch;
  C += (long long)bz * cBatch;
  const int wave  = tid >> 5, lane = tid & 31;
  const int waveM = wave >> 2, waveN = wave & 3;   // 2 x 4 wave grid
  const int lrow  = lane & 15;
  const int khalf = (lane >> 4) << 3;              // lanes 0-15 -> K 0..7/16..23, 16-31 -> 8..15/24..31

  // per-thread staging slots: 512 chunks of 8 halves cover a 128x32 tile
  const int r0 = tid >> 2,           c0 = (tid & 3) << 3;
  const int r1 = (tid + 256) >> 2,   c1 = ((tid + 256) & 3) << 3;
  const bf16* gA0 = A + (long long)(bm * 128 + r0) * lda + c0;
  const bf16* gA1 = A + (long long)(bm * 128 + r1) * lda + c1;
  const bf16* gW0 = W + (long long)(bn * 128 + r0) * ldw + c0;
  const bf16* gW1 = W + (long long)(bn * 128 + r1) * ldw + c1;
  const int la0 = r0 * 40 + c0, la1 = r1 * 40 + c1;

  auto issue = [&](int k0, int buf) {
    async_copy_b128(gA0 + k0, &As[buf][la0]);
    async_copy_b128(gA1 + k0, &As[buf][la1]);
    async_copy_b128(gW0 + k0, &Ws[buf][la0]);
    async_copy_b128(gW1 + k0, &Ws[buf][la1]);
  };

  v8f acc[4][2] = {};

  issue(0, 0);
  wait_async0();
  __syncthreads();

  int buf = 0;
  for (int k0 = 0; k0 < K; k0 += 32) {
    if (k0 + 32 < K) issue(k0 + 32, buf ^ 1);   // prefetch next tile into other buffer

    v16bf af[4], wf[2];
    #pragma unroll
    for (int mf = 0; mf < 4; ++mf) {
      const bf16* p = &As[buf][(waveM * 64 + mf * 16 + lrow) * 40 + khalf];
      v8bf lo = *(const v8bf*)p;
      v8bf hi = *(const v8bf*)(p + 16);
      af[mf] = __builtin_shufflevector(lo, hi, 0,1,2,3,4,5,6,7,8,9,10,11,12,13,14,15);
    }
    #pragma unroll
    for (int nf = 0; nf < 2; ++nf) {
      const bf16* p = &Ws[buf][(waveN * 32 + nf * 16 + lrow) * 40 + khalf];
      v8bf lo = *(const v8bf*)p;
      v8bf hi = *(const v8bf*)(p + 16);
      wf[nf] = __builtin_shufflevector(lo, hi, 0,1,2,3,4,5,6,7,8,9,10,11,12,13,14,15);
    }
    #pragma unroll
    for (int mf = 0; mf < 4; ++mf)
      #pragma unroll
      for (int nf = 0; nf < 2; ++nf)
        acc[mf][nf] = __builtin_amdgcn_wmma_f32_16x16x32_bf16(
            false, af[mf], false, wf[nf], (short)0, acc[mf][nf], false, false);

    wait_async0();        // next buffer's async writes have landed (this wave)
    __syncthreads();      // all waves done reading 'buf' and done filling buf^1
    buf ^= 1;
  }

  const int rbase = (lane >> 4) << 3;
  #pragma unroll
  for (int mf = 0; mf < 4; ++mf) {
    #pragma unroll
    for (int nf = 0; nf < 2; ++nf) {
      const int n = bn * 128 + waveN * 32 + nf * 16 + (lane & 15);
      float bv = (EPI & EPI_BIAS) ? bias[n] : 0.0f;
      float mv = (EPI & EPI_MASK) ? mask[n] : 0.0f;
      #pragma unroll
      for (int r = 0; r < 8; ++r) {
        const int m = bm * 128 + waveM * 64 + mf * 16 + rbase + r;
        float v = acc[mf][nf][r];
        if (EPI & EPI_MASK) v = v * scale + mv;
        v += bv;
        if (EPI & EPI_RES)  v += res[(long long)m * ldres + n];
        if (EPI & EPI_GELU) v = 0.5f * v * (1.0f + erff(v * 0.70710678118654752f));
        C[(long long)m * ldc + n] = (OutT)v;
      }
    }
  }
}

// ---------------- per-head V transpose: Vt[h][d][t] = Vv[t][h*HD+d] ----------------
__global__ void transpose_v_kernel(const bf16* __restrict__ Vv, bf16* __restrict__ Vt) {
  __shared__ bf16 tile[32][33];
  const int h  = blockIdx.z;
  const int t0 = blockIdx.x * 32;
  const int d0 = blockIdx.y * 32;
  #pragma unroll
  for (int i = 0; i < 4; ++i) {
    int tt = threadIdx.y + i * 8;
    tile[tt][threadIdx.x] = Vv[(long long)(t0 + tt) * D_ + h * HD_ + d0 + threadIdx.x];
  }
  __syncthreads();
  #pragma unroll
  for (int i = 0; i < 4; ++i) {
    int dd = threadIdx.y + i * 8;
    Vt[((long long)h * HD_ + d0 + dd) * LT_ + t0 + threadIdx.x] = tile[threadIdx.x][dd];
  }
}

// ---------------- in-place row softmax over 1024 bf16 values ----------------
__global__ __launch_bounds__(256) void softmax_row_bf16(bf16* __restrict__ buf) {
  const long long row = blockIdx.x;
  bf16* p = buf + row * LT_;
  const int t = threadIdx.x;
  float v[4];
  #pragma unroll
  for (int i = 0; i < 4; ++i) v[i] = (float)p[t + i * 256];
  __shared__ float red[256];
  float m = fmaxf(fmaxf(v[0], v[1]), fmaxf(v[2], v[3]));
  red[t] = m; __syncthreads();
  for (int s = 128; s > 0; s >>= 1) { if (t < s) red[t] = fmaxf(red[t], red[t + s]); __syncthreads(); }
  m = red[0]; __syncthreads();
  float e[4], sum = 0.f;
  #pragma unroll
  for (int i = 0; i < 4; ++i) { e[i] = expf(v[i] - m); sum += e[i]; }
  red[t] = sum; __syncthreads();
  for (int s = 128; s > 0; s >>= 1) { if (t < s) red[t] += red[t + s]; __syncthreads(); }
  float inv = 1.0f / red[0];
  #pragma unroll
  for (int i = 0; i < 4; ++i) p[t + i * 256] = (bf16)(e[i] * inv);
}

// ---------------- layernorm over D=4096, fp32 in -> bf16 out ----------------
__global__ __launch_bounds__(256) void layernorm_kernel(const float* __restrict__ x,
                                                        bf16* __restrict__ xn,
                                                        const float* __restrict__ w,
                                                        const float* __restrict__ b) {
  const long long row = blockIdx.x;
  const float* px = x + row * D_;
  bf16* po = xn + row * D_;
  const int t = threadIdx.x;
  float v[16], s = 0.f, ss = 0.f;
  #pragma unroll
  for (int i = 0; i < 16; ++i) { v[i] = px[t + i * 256]; s += v[i]; ss += v[i] * v[i]; }
  __shared__ float r1[256], r2[256];
  r1[t] = s; r2[t] = ss; __syncthreads();
  for (int st = 128; st > 0; st >>= 1) {
    if (t < st) { r1[t] += r1[t + st]; r2[t] += r2[t + st]; }
    __syncthreads();
  }
  float mean = r1[0] * (1.0f / D_);
  float var  = r2[0] * (1.0f / D_) - mean * mean;
  float rstd = rsqrtf(var + 1e-5f);
  #pragma unroll
  for (int i = 0; i < 16; ++i) {
    int c = t + i * 256;
    po[c] = (bf16)((v[i] - mean) * rstd * w[c] + b[c]);
  }
}

// ---------------- score GEMV: h(row,8192) . W2 + b2 + noise*0.07 ----------------
__global__ __launch_bounds__(256) void score_kernel(const bf16* __restrict__ hbuf,
                                                    const float* __restrict__ W2,
                                                    const float* __restrict__ b2,
                                                    const float* __restrict__ noise,
                                                    float* __restrict__ scores) {
  const long long row = blockIdx.x;
  const bf16* p = hbuf + row * MLPH_;
  const int t = threadIdx.x;
  float acc = 0.f;
  for (int j = t; j < MLPH_; j += 256) acc += (float)p[j] * W2[j];
  __shared__ float red[256];
  red[t] = acc; __syncthreads();
  for (int s = 128; s > 0; s >>= 1) { if (t < s) red[t] += red[t + s]; __syncthreads(); }
  if (t == 0) scores[row] = red[0] + b2[0] + noise[row] * 0.07f;
}

// ---------------- top-p: softmax(T=0.05), descending bitonic sort, scan, keep ----------------
__global__ __launch_bounds__(1024) void topp_kernel(const float* __restrict__ scores,
                                                    int* __restrict__ order,
                                                    int* __restrict__ keep) {
  __shared__ float pv[2048];
  __shared__ int   pidx[2048];
  __shared__ float sc[2048];
  const int t = threadIdx.x;
  float s0 = scores[t] * 20.0f;            // / TEMP(0.05)
  float s1 = scores[t + 1024] * 20.0f;
  sc[t] = fmaxf(s0, s1); __syncthreads();
  for (int s = 512; s > 0; s >>= 1) { if (t < s) sc[t] = fmaxf(sc[t], sc[t + s]); __syncthreads(); }
  float m = sc[0]; __syncthreads();
  float e0 = expf(s0 - m), e1 = expf(s1 - m);
  sc[t] = e0 + e1; __syncthreads();
  for (int s = 512; s > 0; s >>= 1) { if (t < s) sc[t] += sc[t + s]; __syncthreads(); }
  float inv = 1.0f / sc[0]; __syncthreads();
  pv[t] = e0 * inv;           pidx[t] = t;
  pv[t + 1024] = e1 * inv;    pidx[t + 1024] = t + 1024;
  __syncthreads();
  // bitonic sort, final order descending by p
  for (int k = 2; k <= 2048; k <<= 1) {
    for (int j = k >> 1; j > 0; j >>= 1) {
      #pragma unroll
      for (int ii = 0; ii < 2; ++ii) {
        int i = t + ii * 1024;
        int ixj = i ^ j;
        if (ixj > i) {
          bool desc = ((i & k) == 0);
          float a = pv[i], b = pv[ixj];
          bool sw = desc ? (a < b) : (a > b);
          if (sw) {
            pv[i] = b; pv[ixj] = a;
            int tmp = pidx[i]; pidx[i] = pidx[ixj]; pidx[ixj] = tmp;
          }
        }
      }
      __syncthreads();
    }
  }
  // inclusive scan of sorted probabilities
  sc[t] = pv[t]; sc[t + 1024] = pv[t + 1024]; __syncthreads();
  for (int off = 1; off < 2048; off <<= 1) {
    float a0 = sc[t];
    float a1 = sc[t + 1024];
    float b0 = (t >= off) ? sc[t - off] : 0.f;
    float b1 = (t + 1024 >= off) ? sc[t + 1024 - off] : 0.f;
    __syncthreads();
    sc[t] = a0 + b0; sc[t + 1024] = a1 + b1;
    __syncthreads();
  }
  order[t] = pidx[t];               order[t + 1024] = pidx[t + 1024];
  keep[t] = (sc[t] <= 0.6f) ? 1 : 0;
  keep[t + 1024] = (sc[t + 1024] <= 0.6f) ? 1 : 0;
}

// ---------------- masked gather of vision rows ----------------
__global__ __launch_bounds__(256) void gather_kernel(const float* __restrict__ vision,
                                                     const int* __restrict__ order,
                                                     const int* __restrict__ keep,
                                                     float* __restrict__ out) {
  const long long row = blockIdx.x;
  const int o  = order[row];
  const int kf = keep[row];
  const float4* src = (const float4*)(vision + (long long)o * D_);
  float4* dst = (float4*)(out + row * (long long)D_);
  const float4 z = make_float4(0.f, 0.f, 0.f, 0.f);
  for (int j = threadIdx.x; j < D_ / 4; j += 256) dst[j] = kf ? src[j] : z;
}

extern "C" void kernel_launch(void* const* d_in, const int* in_sizes, int n_in,
                              void* d_out, int out_size, void* d_ws, size_t ws_size,
                              hipStream_t stream) {
  const float* vision = (const float*)d_in[0];
  const float* text   = (const float*)d_in[1];
  const float* amask  = (const float*)d_in[2];
  const float* noise  = (const float*)d_in[3];
  const float* Wq = (const float*)d_in[4];   const float* bq = (const float*)d_in[5];
  const float* Wk = (const float*)d_in[6];   const float* bk = (const float*)d_in[7];
  const float* Wv = (const float*)d_in[8];   const float* bv = (const float*)d_in[9];
  const float* Wo = (const float*)d_in[10];  const float* bo = (const float*)d_in[11];
  const float* lnw = (const float*)d_in[12]; const float* lnb = (const float*)d_in[13];
  const float* W1 = (const float*)d_in[14];  const float* b1 = (const float*)d_in[15];
  const float* W2 = (const float*)d_in[16];  const float* b2 = (const float*)d_in[17];
  float* out = (float*)d_out;

  size_t off = 0;
  auto take = [&](size_t bytes) -> void* {
    void* p = (char*)d_ws + off;
    off += (bytes + 255) & ~(size_t)255;
    return p;
  };
  bf16* Vb  = (bf16*)take((size_t)LV_ * D_ * 2);
  bf16* Tb  = (bf16*)take((size_t)LT_ * D_ * 2);
  bf16* Wb  = (bf16*)take((size_t)D_ * D_ * 2);            // reused: Wq/Wk/Wv/Wo bf16
  bf16* Qb  = (bf16*)take((size_t)LV_ * D_ * 2);
  bf16* Kb  = (bf16*)take((size_t)LT_ * D_ * 2);
  bf16* Vvb = (bf16*)take((size_t)LT_ * D_ * 2);
  bf16* Vt  = (bf16*)take((size_t)H_ * HD_ * LT_ * 2);
  bf16* Pb  = (bf16*)take((size_t)H_ * LV_ * LT_ * 2);     // logits/probs; later reused as bf16 W1
  bf16* Ctx = (bf16*)take((size_t)LV_ * D_ * 2);
  float* X  = (float*)take((size_t)LV_ * D_ * 4);
  bf16* Xn  = (bf16*)take((size_t)LV_ * D_ * 2);
  bf16* Hb  = (bf16*)take((size_t)LV_ * MLPH_ * 2);
  float* Sc = (float*)take((size_t)LV_ * 4);
  int* Ord  = (int*)take((size_t)LV_ * 4);
  int* Keep = (int*)take((size_t)LV_ * 4);
  (void)ws_size; (void)in_sizes; (void)n_in; (void)out_size;

  const long long nv = (long long)LV_ * D_;
  const long long nt = (long long)LT_ * D_;
  const long long nw = (long long)D_ * D_;
  const long long n1 = (long long)MLPH_ * D_;

  cvt_f32_bf16<<<dim3((unsigned)(nv / 1024)), 256, 0, stream>>>(vision, Vb, nv);
  cvt_f32_bf16<<<dim3((unsigned)(nt / 1024)), 256, 0, stream>>>(text, Tb, nt);

  // Q = Vb * Wq^T + bq  -> bf16
  cvt_f32_bf16<<<dim3((unsigned)(nw / 1024)), 256, 0, stream>>>(Wq, Wb, nw);
  gemm_bf16_kernel<EPI_BIAS, bf16><<<dim3(D_ / 128, LV_ / 128, 1), 256, 0, stream>>>(
      Vb, Wb, Qb, LV_, D_, D_, D_, D_, D_, 0, 0, 0, bq, nullptr, 0, nullptr, 1.0f);
  // K = Tb * Wk^T + bk  -> bf16
  cvt_f32_bf16<<<dim3((unsigned)(nw / 1024)), 256, 0, stream>>>(Wk, Wb, nw);
  gemm_bf16_kernel<EPI_BIAS, bf16><<<dim3(D_ / 128, LT_ / 128, 1), 256, 0, stream>>>(
      Tb, Wb, Kb, LT_, D_, D_, D_, D_, D_, 0, 0, 0, bk, nullptr, 0, nullptr, 1.0f);
  // V = Tb * Wv^T + bv  -> bf16
  cvt_f32_bf16<<<dim3((unsigned)(nw / 1024)), 256, 0, stream>>>(Wv, Wb, nw);
  gemm_bf16_kernel<EPI_BIAS, bf16><<<dim3(D_ / 128, LT_ / 128, 1), 256, 0, stream>>>(
      Tb, Wb, Vvb, LT_, D_, D_, D_, D_, D_, 0, 0, 0, bv, nullptr, 0, nullptr, 1.0f);

  transpose_v_kernel<<<dim3(LT_ / 32, HD_ / 32, H_), dim3(32, 8), 0, stream>>>(Vvb, Vt);

  // logits[h] = (Qh * Kh^T) / 16 + mask  -> bf16, batched over 16 heads
  gemm_bf16_kernel<EPI_MASK, bf16><<<dim3(LT_ / 128, LV_ / 128, H_), 256, 0, stream>>>(
      Qb, Kb, Pb, LV_, LT_, HD_, D_, D_, LT_,
      (long long)HD_, (long long)HD_, (long long)LV_ * LT_,
      nullptr, nullptr, 0, amask, 0.0625f);

  softmax_row_bf16<<<dim3(H_ * LV_), 256, 0, stream>>>(Pb);

  // ctx[h] = P[h] * V[h]  (A*B via W = Vt[h] (HDxLT)), out into head-sliced columns
  gemm_bf16_kernel<0, bf16><<<dim3(HD_ / 128, LV_ / 128, H_), 256, 0, stream>>>(
      Pb, Vt, Ctx, LV_, HD_, LT_, LT_, LT_, D_,
      (long long)LV_ * LT_, (long long)HD_ * LT_, (long long)HD_,
      nullptr, nullptr, 0, nullptr, 1.0f);

  // x = vision + ctx * Wo^T + bo  -> fp32
  cvt_f32_bf16<<<dim3((unsigned)(nw / 1024)), 256, 0, stream>>>(Wo, Wb, nw);
  gemm_bf16_kernel<EPI_BIAS | EPI_RES, float><<<dim3(D_ / 128, LV_ / 128, 1), 256, 0, stream>>>(
      Ctx, Wb, X, LV_, D_, D_, D_, D_, D_, 0, 0, 0, bo, vision, D_, nullptr, 1.0f);

  layernorm_kernel<<<dim3(LV_), 256, 0, stream>>>(X, Xn, lnw, lnb);

  // h = gelu(xn * W1^T + b1) -> bf16 (W1 bf16 staged into the freed logits region)
  cvt_f32_bf16<<<dim3((unsigned)(n1 / 1024)), 256, 0, stream>>>(W1, Pb, n1);
  gemm_bf16_kernel<EPI_BIAS | EPI_GELU, bf16><<<dim3(MLPH_ / 128, LV_ / 128, 1), 256, 0, stream>>>(
      Xn, Pb, Hb, LV_, MLPH_, D_, D_, D_, MLPH_, 0, 0, 0, b1, nullptr, 0, nullptr, 1.0f);

  score_kernel<<<dim3(LV_), 256, 0, stream>>>(Hb, W2, b2, noise, Sc);
  topp_kernel<<<dim3(1), 1024, 0, stream>>>(Sc, Ord, Keep);
  gather_kernel<<<dim3(LV_), 256, 0, stream>>>(vision, Ord, Keep, out);
}
